// DecompositionModel_44143673868641
// MI455X (gfx1250) — matrix operand order repaired and checked by reading, not compiled, not run on hardware
//
#include <hip/hip_runtime.h>
#include <hip/hip_bf16.h>

typedef __attribute__((ext_vector_type(16))) __bf16 v16bf;
typedef __attribute__((ext_vector_type(8)))  float  v8f;

#define LDA 40    // padded A tile stride (halves), 80B rows (16B aligned)
#define LDB 136   // padded B tile stride (halves), 272B rows (16B aligned)

// ---- CDNA5 async global->LDS DMA (ASYNCcnt-tracked, bypasses VGPRs) ----
__device__ __forceinline__ void async_g2l_b128(unsigned lds_addr, const void* gaddr) {
  asm volatile("global_load_async_to_lds_b128 %0, %1, off"
               :: "v"(lds_addr), "v"(gaddr) : "memory");
}
__device__ __forceinline__ void wait_async0() {
  asm volatile("s_wait_asynccnt 0x0" ::: "memory");
}

// ============================================================
// Kernel 1: b path — x_feat 3x3 conv, window(2,2), LN, b_proj, l2norm, RoPE
// Outputs bws/brotws: (4 win, 16 head, 16 k, 4 d) fp32
// ============================================================
__global__ __launch_bounds__(256) void bpath_kernel(
    const float* __restrict__ x,    // (3,8,8)
    const float* __restrict__ xcw,  // (64,3,3,3)
    const float* __restrict__ xcb,  // (64)
    const float* __restrict__ nbg, const float* __restrict__ nbb,
    const float* __restrict__ bpw, const float* __restrict__ bpb,
    float* __restrict__ bws, float* __restrict__ brotws)
{
  __shared__ float sxf[64 * 64];    // x_feat[c][i*8+j]
  for (int o = threadIdx.x; o < 4096; o += 256) {
    int c = o >> 6, ij = o & 63, i = ij >> 3, j = ij & 7;
    float acc = xcb[c];
    for (int ic = 0; ic < 3; ++ic)
      for (int kh = 0; kh < 3; ++kh)
        for (int kw = 0; kw < 3; ++kw) {
          int iy = i + kh - 1, ix = j + kw - 1;
          if (iy >= 0 && iy < 8 && ix >= 0 && ix < 8)
            acc += x[ic * 64 + iy * 8 + ix] * xcw[((c * 3 + ic) * 3 + kh) * 3 + kw];
        }
    sxf[o] = acc;
  }
  __syncthreads();
  for (int it = threadIdx.x; it < 1024; it += 256) {
    int w = it >> 8, h = (it >> 4) & 15, k = it & 15;
    int m = w >> 1, n = w & 1, hh = k >> 2, ww = k & 3;
    int row = m * 4 + hh, col = n * 4 + ww;
    float v[4];
    #pragma unroll
    for (int d = 0; d < 4; ++d) v[d] = sxf[(h * 4 + d) * 64 + row * 8 + col];
    float mu = 0.25f * (v[0] + v[1] + v[2] + v[3]);
    float var = 0.25f * ((v[0]-mu)*(v[0]-mu) + (v[1]-mu)*(v[1]-mu) +
                         (v[2]-mu)*(v[2]-mu) + (v[3]-mu)*(v[3]-mu));
    float inv = rsqrtf(var + 1e-5f);
    float ln[4], bv[4];
    #pragma unroll
    for (int d = 0; d < 4; ++d) ln[d] = (v[d] - mu) * inv * nbg[d] + nbb[d];
    #pragma unroll
    for (int d = 0; d < 4; ++d) {
      float s = bpb[d];
      #pragma unroll
      for (int e = 0; e < 4; ++e) s += ln[e] * bpw[d * 4 + e];
      bv[d] = s;
      bws[it * 4 + d] = s;
    }
    float nrm = sqrtf(bv[0]*bv[0] + bv[1]*bv[1] + bv[2]*bv[2] + bv[3]*bv[3]);
    float rcp = 1.0f / fmaxf(nrm, 1e-12f);
    float b0 = bv[0]*rcp, b1 = bv[1]*rcp, b2 = bv[2]*rcp, b3 = bv[3]*rcp;
    float s0, c0, s1, c1;
    sincosf((float)hh, &s0, &c0);
    sincosf((float)ww, &s1, &c1);
    brotws[it * 4 + 0] = b0 * c0 - b1 * s0;
    brotws[it * 4 + 1] = b0 * s0 + b1 * c0;
    brotws[it * 4 + 2] = b2 * c1 - b3 * s1;
    brotws[it * 4 + 3] = b2 * s1 + b3 * c1;
  }
}

// ============================================================
// Kernel 2: bilinear resize 8x8 -> 1024x1024 (half-pixel centers, edge clamp)
// ============================================================
__global__ __launch_bounds__(256) void resize_kernel(
    const float* __restrict__ x, float* __restrict__ out)
{
  int tid = blockIdx.x * 256 + threadIdx.x;    // 3 * 1048576 threads
  int c = tid >> 20, idx = tid & 0xFFFFF;
  int oy = idx >> 10, ox = idx & 1023;
  float sy = (oy + 0.5f) * (8.0f / 1024.0f) - 0.5f;
  float sx = (ox + 0.5f) * (8.0f / 1024.0f) - 0.5f;
  int y0 = (int)floorf(sy), x0 = (int)floorf(sx);
  float fy = sy - (float)y0, fx = sx - (float)x0;
  int y0c = min(max(y0, 0), 7), y1c = min(max(y0 + 1, 0), 7);
  int x0c = min(max(x0, 0), 7), x1c = min(max(x0 + 1, 0), 7);
  const float* xc = x + c * 64;
  float v00 = xc[y0c * 8 + x0c], v01 = xc[y0c * 8 + x1c];
  float v10 = xc[y1c * 8 + x0c], v11 = xc[y1c * 8 + x1c];
  out[tid] = (v00 * (1.f - fx) + v01 * fx) * (1.f - fy) +
             (v10 * (1.f - fx) + v11 * fx) * fy;
}

// ============================================================
// Kernel 3: 3x3 conv (3->64 ch) over 1024x1024, output bf16 (WMMA feed)
// ============================================================
__global__ __launch_bounds__(256) void rconv_kernel(
    const float* __restrict__ img,  // (3,1024,1024)  (L2 resident: 12.6MB)
    const float* __restrict__ w,    // (64,3,3,3)
    const float* __restrict__ b,
    __bf16* __restrict__ out)       // (64,1024,1024)
{
  int tid = blockIdx.x * 256 + threadIdx.x;    // 64 * 1048576 threads
  int oc = tid >> 20, idx = tid & 0xFFFFF;
  int yy = idx >> 10, xx = idx & 1023;
  float acc = b[oc];
  #pragma unroll
  for (int ic = 0; ic < 3; ++ic)
    #pragma unroll
    for (int kh = 0; kh < 3; ++kh)
      #pragma unroll
      for (int kw = 0; kw < 3; ++kw) {
        int iy = yy + kh - 1, ix = xx + kw - 1;
        if (iy >= 0 && iy < 1024 && ix >= 0 && ix < 1024)
          acc += img[ic * 1048576 + iy * 1024 + ix] * w[((oc * 3 + ic) * 3 + kh) * 3 + kw];
      }
  out[tid] = (__bf16)acc;
}

// ============================================================
// Kernel 4: fp32 -> bf16 convert
// ============================================================
__global__ __launch_bounds__(256) void cvt_kernel(
    const float* __restrict__ in, __bf16* __restrict__ out, int n)
{
  int i = blockIdx.x * 256 + threadIdx.x;
  if (i < n) out[i] = (__bf16)in[i];
}

// ============================================================
// WMMA fragment loaders per CDNA5 ISA 7.12.2 (wave32):
//  A 16x32 bf16: lane&15 = M row; lane>>4 picks K-half; e[0..7]=K kb..kb+7,
//                e[8..15]=K kb+16..kb+23 (kb = (lane>>4)*8)
//  B 32x16 bf16: lane&15 = N col; lane>>4 picks K-half; e[0..7]=K kb..kb+7,
//                e[8..15]=K kb+16..kb+23 (kb = (lane>>4)*8)
// ============================================================
__device__ __forceinline__ v16bf load_a_frag(const __bf16* As, int rowBase, int lane) {
  union { v16bf v; __bf16 e[16]; } u;
  const __bf16* p = As + (rowBase + (lane & 15)) * LDA + ((lane >> 4) << 3);
  #pragma unroll
  for (int i = 0; i < 8; ++i) { u.e[i] = p[i]; u.e[8 + i] = p[16 + i]; }
  return u.v;
}
__device__ __forceinline__ v16bf load_b_frag(const __bf16* Bs, int nBase, int lane) {
  union { v16bf v; __bf16 e[16]; } u;
  const __bf16* p = Bs + ((lane >> 4) << 3) * LDB + nBase + (lane & 15);
  #pragma unroll
  for (int i = 0; i < 8; ++i) { u.e[i] = p[i * LDB]; u.e[8 + i] = p[(16 + i) * LDB]; }
  return u.v;
}

// ============================================================
// Kernel 5: patch conv as implicit GEMM  C[4096 ch][4096 pos], K=16384
//   A = patch_w bf16, B = im2col of res_x bf16.
//   128x128 WG tile, 8 waves 4(M)x2(N), wave tile 32x64, K-step 32.
//   Double-buffered LDS; A tile moved by global_load_async_to_lds_b128.
// ============================================================
__global__ __launch_bounds__(256) void patch_gemm_kernel(
    const __bf16* __restrict__ A,    // (4096,16384)
    const __bf16* __restrict__ RX,   // (64,1024,1024)
    const float* __restrict__ bias,  // (4096)
    float* __restrict__ C)           // (4096,4096)
{
  __shared__ __bf16 As[2][128 * LDA];
  __shared__ __bf16 Bs[2][32 * LDB];
  const int cbase = blockIdx.y << 7;
  const int pbase = blockIdx.x << 7;
  const int lane = threadIdx.x & 31;
  const int wave = threadIdx.x >> 5;
  const int wm = (wave & 3) << 5;
  const int wn = (wave >> 2) << 6;
  const int basei = threadIdx.x * 16;
  const int arow = basei >> 5, acol = basei & 31;     // A: 16 halves along K
  const int bkk0 = basei & 31, bj = basei >> 5;       // B: 16 halves along K (im2col)
  const int bp = pbase + bj;
  const int boh = bp >> 6, bow = bp & 63;

  v8f acc[2][4];
  #pragma unroll
  for (int mt = 0; mt < 2; ++mt)
    #pragma unroll
    for (int nt = 0; nt < 4; ++nt)
      #pragma unroll
      for (int r = 0; r < 8; ++r) acc[mt][nt][r] = 0.0f;

  const __bf16* gA = A + (cbase + arow) * 16384 + acol;

  // prologue: fill buffer 0
  {
    unsigned sp = (unsigned)(size_t)&As[0][arow * LDA + acol];
    async_g2l_b128(sp, gA);
    async_g2l_b128(sp + 16u, gA + 8);
    int kb = bkk0;                                    // k0 = 0
    int ci = kb >> 8, kh = (kb >> 4) & 15;
    const __bf16* gp = RX + (ci << 20) + (boh * 16 + kh) * 1024 + bow * 16;
    __bf16 tmp[16];
    #pragma unroll
    for (int i = 0; i < 16; ++i) tmp[i] = gp[i];
    #pragma unroll
    for (int i = 0; i < 16; ++i) Bs[0][(bkk0 + i) * LDB + bj] = tmp[i];
  }

  int buf = 0;
  for (int k0 = 0; k0 < 16384; k0 += 32) {
    wait_async0();
    __syncthreads();
    if (k0 + 32 < 16384) {                            // prefetch next tiles
      int nk = k0 + 32;
      unsigned sp = (unsigned)(size_t)&As[buf ^ 1][arow * LDA + acol];
      async_g2l_b128(sp, gA + nk);
      async_g2l_b128(sp + 16u, gA + nk + 8);
      int kb = nk + bkk0;
      int ci = kb >> 8, kh = (kb >> 4) & 15;
      const __bf16* gp = RX + (ci << 20) + (boh * 16 + kh) * 1024 + bow * 16;
      __bf16 tmp[16];
      #pragma unroll
      for (int i = 0; i < 16; ++i) tmp[i] = gp[i];
      #pragma unroll
      for (int i = 0; i < 16; ++i) Bs[buf ^ 1][(bkk0 + i) * LDB + bj] = tmp[i];
    }
    v16bf af[2], bfr[4];
    af[0] = load_a_frag(&As[buf][0], wm, lane);
    af[1] = load_a_frag(&As[buf][0], wm + 16, lane);
    #pragma unroll
    for (int nt = 0; nt < 4; ++nt) bfr[nt] = load_b_frag(&Bs[buf][0], wn + (nt << 4), lane);
    #pragma unroll
    for (int mt = 0; mt < 2; ++mt)
      #pragma unroll
      for (int nt = 0; nt < 4; ++nt)
        acc[mt][nt] = __builtin_amdgcn_wmma_f32_16x16x32_bf16(
            false, af[mt], false, bfr[nt], (short)0, acc[mt][nt], false, false);
    buf ^= 1;
  }
  // store: C/D layout -> VGPR r holds M=r (lanes 0-15) / M=r+8 (lanes 16-31)
  const int mr = (lane >> 4) << 3;
  const int nc = lane & 15;
  #pragma unroll
  for (int mt = 0; mt < 2; ++mt)
    #pragma unroll
    for (int nt = 0; nt < 4; ++nt)
      #pragma unroll
      for (int r = 0; r < 8; ++r) {
        int c = cbase + wm + (mt << 4) + mr + r;
        int p = pbase + wn + (nt << 4) + nc;
        C[c * 4096 + p] = acc[mt][nt][r] + bias[c];
      }
}

// ============================================================
// Kernel 6: attention path. 1 thread per (window, head, q).
// ============================================================
__global__ __launch_bounds__(256) void attn_kernel(
    const float* __restrict__ resbase,   // (4096,4096)
    const float* __restrict__ bws, const float* __restrict__ brotws,
    const float* __restrict__ qpw, const float* __restrict__ qpb,
    const float* __restrict__ nqg, const float* __restrict__ nqb,
    const float* __restrict__ qaw, const float* __restrict__ qab,
    const float* __restrict__ pos,       // (16,65536,4)
    float* __restrict__ attn_out,        // (4,16,65536,16)
    __bf16* __restrict__ Y)              // (4096,4096)
{
  __shared__ float sb[64], sbr[64], sqw[16], sqb[4], sng[4], snb[4], saw[4], sab[1];
  const int w = blockIdx.x >> 12;
  const int h = (blockIdx.x >> 8) & 15;
  const int qidx = ((blockIdx.x & 255) << 8) | threadIdx.x;
  const int t = threadIdx.x;
  if (t < 64) { int base = (w * 16 + h) << 6; sb[t] = bws[base + t]; sbr[t] = brotws[base + t]; }
  else if (t < 80)  sqw[t - 64] = qpw[t - 64];
  else if (t < 84)  sqb[t - 80] = qpb[t - 80];
  else if (t < 88)  sng[t - 84] = nqg[t - 84];
  else if (t < 92)  snb[t - 88] = nqb[t - 88];
  else if (t < 96)  saw[t - 92] = qaw[t - 92];
  else if (t == 96) sab[0] = qab[0];
  __syncthreads();

  const int m = w >> 1, n = w & 1;
  const int qrow = qidx >> 8, qcol = qidx & 255;
  const int H512 = (m << 8) + qrow, W512 = (n << 8) + qcol;
  const int r1 = H512 >> 3, hnr = H512 & 7;
  const int r2 = W512 >> 3, wnr = W512 & 7;
  const int psp = r1 * 64 + r2;
  const int cb = hnr * 8 + wnr;

  float rwv[4];
  #pragma unroll
  for (int d = 0; d < 4; ++d)
    rwv[d] = resbase[(((h << 2) + d) * 64 + cb) * 4096 + psp];

  float mu = 0.25f * (rwv[0] + rwv[1] + rwv[2] + rwv[3]);
  float var = 0.25f * ((rwv[0]-mu)*(rwv[0]-mu) + (rwv[1]-mu)*(rwv[1]-mu) +
                       (rwv[2]-mu)*(rwv[2]-mu) + (rwv[3]-mu)*(rwv[3]-mu));
  float inv = rsqrtf(var + 1e-5f);
  float ln[4], q[4];
  #pragma unroll
  for (int d = 0; d < 4; ++d) ln[d] = (rwv[d] - mu) * inv * sng[d] + snb[d];
  #pragma unroll
  for (int d = 0; d < 4; ++d) {
    float s = sqb[d];
    #pragma unroll
    for (int e = 0; e < 4; ++e) s += ln[e] * sqw[d * 4 + e];
    q[d] = s;
  }
  float g = sab[0] + q[0]*saw[0] + q[1]*saw[1] + q[2]*saw[2] + q[3]*saw[3];
  g = 1.0f / (1.0f + __expf(-g));
  const float4 pv = *(const float4*)(pos + (((h << 16) | qidx) << 2));
  float q0 = q[0]*g + pv.x*(1.f-g), q1 = q[1]*g + pv.y*(1.f-g);
  float q2 = q[2]*g + pv.z*(1.f-g), q3 = q[3]*g + pv.w*(1.f-g);
  float nn = sqrtf(q0*q0 + q1*q1 + q2*q2 + q3*q3);
  float rcp = 1.0f / fmaxf(nn, 1e-12f);
  q0 *= rcp; q1 *= rcp; q2 *= rcp; q3 *= rcp;
  float s0, c0, s1, c1;
  sincosf((float)qrow, &s0, &c0);
  sincosf((float)qcol, &s1, &c1);
  const float qr0 = q0 * c0 - q1 * s0, qr1 = q0 * s0 + q1 * c0;
  const float qr2 = q2 * c1 - q3 * s1, qr3 = q2 * s1 + q3 * c1;

  float av[16], o0 = 0.f, o1 = 0.f, o2 = 0.f, o3 = 0.f;
  #pragma unroll
  for (int k = 0; k < 16; ++k) {
    float a = qr0*sbr[k*4+0] + qr1*sbr[k*4+1] + qr2*sbr[k*4+2] + qr3*sbr[k*4+3];
    av[k] = a;
    o0 += a * sb[k*4+0]; o1 += a * sb[k*4+1];
    o2 += a * sb[k*4+2]; o3 += a * sb[k*4+3];
  }
  float4* ap = (float4*)(attn_out + ((((w << 4) | h) * 65536 + qidx) << 4));
  ap[0] = make_float4(av[0],  av[1],  av[2],  av[3]);
  ap[1] = make_float4(av[4],  av[5],  av[6],  av[7]);
  ap[2] = make_float4(av[8],  av[9],  av[10], av[11]);
  ap[3] = make_float4(av[12], av[13], av[14], av[15]);
  float ov[4] = {o0, o1, o2, o3};
  #pragma unroll
  for (int d = 0; d < 4; ++d)
    Y[(((h << 2) + d) * 64 + cb) * 4096 + psp] = (__bf16)(ov[d] + rwv[d]);
}

// ============================================================
// Kernel 7: 1x1 proj conv as GEMM, K=4096. Double-buffered; BOTH tiles
//   moved with global_load_async_to_lds_b128 (all-contiguous copies).
// ============================================================
__global__ __launch_bounds__(256) void proj_gemm_kernel(
    const __bf16* __restrict__ A,    // proj_w bf16 (4096,4096)
    const __bf16* __restrict__ B,    // y bf16 (4096,4096)
    const float* __restrict__ bias,  // proj_b
    float* __restrict__ C)           // x_final (4096,4096)
{
  __shared__ __bf16 As[2][128 * LDA];
  __shared__ __bf16 Bs[2][32 * LDB];
  const int cbase = blockIdx.y << 7;
  const int pbase = blockIdx.x << 7;
  const int lane = threadIdx.x & 31;
  const int wave = threadIdx.x >> 5;
  const int wm = (wave & 3) << 5;
  const int wn = (wave >> 2) << 6;
  const int basei = threadIdx.x * 16;
  const int arow = basei >> 5, acol = basei & 31;
  const int bkk = basei >> 7, bj0 = basei & 127;

  v8f acc[2][4];
  #pragma unroll
  for (int mt = 0; mt < 2; ++mt)
    #pragma unroll
    for (int nt = 0; nt < 4; ++nt)
      #pragma unroll
      for (int r = 0; r < 8; ++r) acc[mt][nt][r] = 0.0f;

  const __bf16* gA = A + (cbase + arow) * 4096 + acol;
  const __bf16* gB = B + bkk * 4096 + pbase + bj0;

  auto issue = [&](int bufsel, int k0) {
    unsigned spA = (unsigned)(size_t)&As[bufsel][arow * LDA + acol];
    async_g2l_b128(spA, gA + k0);
    async_g2l_b128(spA + 16u, gA + k0 + 8);
    unsigned spB = (unsigned)(size_t)&Bs[bufsel][bkk * LDB + bj0];
    async_g2l_b128(spB, gB + k0 * 4096);
    async_g2l_b128(spB + 16u, gB + k0 * 4096 + 8);
  };

  issue(0, 0);
  int buf = 0;
  for (int k0 = 0; k0 < 4096; k0 += 32) {
    wait_async0();
    __syncthreads();
    if (k0 + 32 < 4096) issue(buf ^ 1, k0 + 32);
    v16bf af[2], bfr[4];
    af[0] = load_a_frag(&As[buf][0], wm, lane);
    af[1] = load_a_frag(&As[buf][0], wm + 16, lane);
    #pragma unroll
    for (int nt = 0; nt < 4; ++nt) bfr[nt] = load_b_frag(&Bs[buf][0], wn + (nt << 4), lane);
    #pragma unroll
    for (int mt = 0; mt < 2; ++mt)
      #pragma unroll
      for (int nt = 0; nt < 4; ++nt)
        acc[mt][nt] = __builtin_amdgcn_wmma_f32_16x16x32_bf16(
            false, af[mt], false, bfr[nt], (short)0, acc[mt][nt], false, false);
    buf ^= 1;
  }
  const int mr = (lane >> 4) << 3;
  const int nc = lane & 15;
  #pragma unroll
  for (int mt = 0; mt < 2; ++mt)
    #pragma unroll
    for (int nt = 0; nt < 4; ++nt)
      #pragma unroll
      for (int r = 0; r < 8; ++r) {
        int c = cbase + wm + (mt << 4) + mr + r;
        int p = pbase + wn + (nt << 4) + nc;
        C[c * 4096 + p] = acc[mt][nt][r] + bias[c];
      }
}

// ============================================================
// Launch
// ============================================================
extern "C" void kernel_launch(void* const* d_in, const int* in_sizes, int n_in,
                              void* d_out, int out_size, void* d_ws, size_t ws_size,
                              hipStream_t stream) {
  (void)in_sizes; (void)n_in; (void)out_size; (void)ws_size;
  const float* x      = (const float*)d_in[0];
  const float* xcw    = (const float*)d_in[1];
  const float* xcb    = (const float*)d_in[2];
  const float* rcw    = (const float*)d_in[3];
  const float* rcb    = (const float*)d_in[4];
  const float* patchw = (const float*)d_in[5];
  const float* patchb = (const float*)d_in[6];
  const float* bpw    = (const float*)d_in[7];
  const float* bpb    = (const float*)d_in[8];
  const float* qpw    = (const float*)d_in[9];
  const float* qpb    = (const float*)d_in[10];
  const float* nbg    = (const float*)d_in[11];
  const float* nbb    = (const float*)d_in[12];
  const float* nqg    = (const float*)d_in[13];
  const float* nqb    = (const float*)d_in[14];
  const float* qaw    = (const float*)d_in[15];
  const float* qab    = (const float*)d_in[16];
  const float* pos    = (const float*)d_in[17];
  const float* projw  = (const float*)d_in[18];
  const float* projb  = (const float*)d_in[19];

  float* out_final = (float*)d_out;            // (4096,64,64) = 16777216 f32
  float* out_attn  = (float*)d_out + 16777216; // (4,16,65536,16) f32

  char* ws = (char*)d_ws;
  float*  resized = (float*)(ws + 0);              // 12,582,912 B
  __bf16* resx    = (__bf16*)(ws + 12582912ull);   // 134,217,728 B
  float*  resbase = (float*)(ws + 146800640ull);   //  67,108,864 B
  __bf16* patchwb = (__bf16*)(ws + 213909504ull);  // 134,217,728 B
  __bf16* projwb  = (__bf16*)(ws + 348127232ull);  //  33,554,432 B
  __bf16* ybf     = (__bf16*)(ws + 381681664ull);  //  33,554,432 B
  float*  bws     = (float*)(ws + 415236096ull);   //      16,384 B
  float*  brotws  = (float*)(ws + 415252480ull);   //      16,384 B

  bpath_kernel<<<1, 256, 0, stream>>>(x, xcw, xcb, nbg, nbb, bpw, bpb, bws, brotws);
  resize_kernel<<<12288, 256, 0, stream>>>(x, resized);
  rconv_kernel<<<262144, 256, 0, stream>>>(resized, rcw, rcb, resx);
  cvt_kernel<<<262144, 256, 0, stream>>>(patchw, patchwb, 67108864);
  cvt_kernel<<<65536, 256, 0, stream>>>(projw, projwb, 16777216);
  patch_gemm_kernel<<<dim3(32, 32), 256, 0, stream>>>(patchwb, resx, patchb, resbase);
  attn_kernel<<<16384, 256, 0, stream>>>(resbase, bws, brotws, qpw, qpb, nqg, nqb,
                                         qaw, qab, pos, out_attn, ybf);
  proj_gemm_kernel<<<dim3(32, 32), 256, 0, stream>>>(projwb, ybf, projb, out_final);
}
